// NeuralODE_16922171146296
// MI455X (gfx1250) — matrix operand order repaired and checked by reading, not compiled, not run on hardware
//
#include <hip/hip_runtime.h>
#include <math.h>

#define BATCH  2048
#define TSEQ   64
#define DIN    4
#define MMETA  4
#define H      256
#define NSTEPS 32
#define TILE_B 16
#define BLD    264   // padded bf16 LDS row stride (256 + 8)

typedef __attribute__((ext_vector_type(16))) __bf16 v16bf;
typedef __attribute__((ext_vector_type(8)))  __bf16 v8bf;
typedef __attribute__((ext_vector_type(8)))  float  v8f;

// ---- Dormand-Prince (dopri5) coefficients ----
constexpr float A21c = 1.0f/5.0f;
constexpr float A31c = 3.0f/40.0f,        A32c = 9.0f/40.0f;
constexpr float A41c = 44.0f/45.0f,       A42c = -56.0f/15.0f,     A43c = 32.0f/9.0f;
constexpr float A51c = 19372.0f/6561.0f,  A52c = -25360.0f/2187.0f,
                A53c = 64448.0f/6561.0f,  A54c = -212.0f/729.0f;
constexpr float A61c = 9017.0f/3168.0f,   A62c = -355.0f/33.0f,
                A63c = 46732.0f/5247.0f,  A64c = 49.0f/176.0f,     A65c = -5103.0f/18656.0f;
constexpr float B1c  = 35.0f/384.0f,      B3c  = 500.0f/1113.0f,   B4c  = 125.0f/192.0f,
                B5c  = -2187.0f/6784.0f,  B6c  = 11.0f/84.0f;

// ------------------------------------------------------------------
// packed bf16 weight arena (device global; rewritten every launch)
// layout per matrix: fragment-major, per fragment 32 lanes x 16 bf16,
// value[frag][lane][i] = W[nt*16 + (lane&15)][kt*32 + ((lane&16)?16:0) + i]
// ------------------------------------------------------------------
constexpr int SZ_WIH = 768 * 32;        // K padded 8 -> 32
constexpr int SZ_WHH = 768 * 256;
constexpr int SZ_EW1 = 256 * 256;
constexpr int SZ_EW2 = 512 * 256;
constexpr int SZ_OW  = 256 * 256;
constexpr int OFF_WIH = 0;
constexpr int OFF_WHH = OFF_WIH + SZ_WIH;
constexpr int OFF_EW1 = OFF_WHH + SZ_WHH;
constexpr int OFF_EW2 = OFF_EW1 + SZ_EW1;
constexpr int OFF_OW1 = OFF_EW2 + SZ_EW2;
constexpr int OFF_OW2 = OFF_OW1 + SZ_OW;
constexpr int OFF_OW3 = OFF_OW2 + SZ_OW;
constexpr int OFF_OW4 = OFF_OW3 + SZ_OW;
constexpr int PACK_TOTAL = OFF_OW4 + SZ_OW;   // 679,936 bf16 = ~1.30 MB

__device__ __bf16 g_wpack[PACK_TOTAL];

__global__ __launch_bounds__(256)
void pack_weights(const float* __restrict__ W, int dstOff, int N, int K, int Kreal) {
  __bf16* out = g_wpack + dstOff;
  const int KT = K >> 5;
  const int total = N * K;
  for (int p = blockIdx.x * blockDim.x + threadIdx.x; p < total;
       p += gridDim.x * blockDim.x) {
    const int i    = p & 15;
    const int l    = (p >> 4) & 31;
    const int frag = p >> 9;
    const int kt   = frag % KT;
    const int nt   = frag / KT;
    const int n    = nt * 16 + (l & 15);
    const int k    = kt * 32 + ((l & 16) ? 16 : 0) + i;
    const float v  = (k < Kreal) ? W[(size_t)n * Kreal + k] : 0.0f;
    out[p] = (__bf16)v;
  }
}

// ------------------------------------------------------------------
// WMMA helpers (CDNA5 wave32, D = A(16x32) * B(32x16) + C, fp32 acc)
// ------------------------------------------------------------------
__device__ __forceinline__ v8f wmma_bf16(v16bf a, v16bf b, v8f c) {
  return __builtin_amdgcn_wmma_f32_16x16x32_bf16(false, a, false, b, (short)0, c,
                                                 false, false);
}

__device__ __forceinline__ v16bf join16(v8bf lo, v8bf hi) {
  v16bf f;
#pragma unroll
  for (int i = 0; i < 8; ++i) { f[i] = lo[i]; f[i + 8] = hi[i]; }
  return f;
}

// A-matrix fragment from LDS (16xK bf16, row M = lane&15): two 16B ds loads.
__device__ __forceinline__ v16bf frag_A_lds(const __bf16* S, int ld, int kb, int lane) {
  const __bf16* p = S + (lane & 15) * ld + kb + ((lane & 16) ? 16 : 0);
  const v8bf lo = *(const v8bf*)(p);
  const v8bf hi = *(const v8bf*)(p + 8);
  return join16(lo, hi);
}

// B-matrix fragment from packed arena: two contiguous 16B global loads.
__device__ __forceinline__ v16bf frag_B_pk(const __bf16* base, int KT, int n0, int k,
                                           int lane) {
  const __bf16* p = base + ((size_t)((n0 >> 4) * KT + (k >> 5)) * 32 + lane) * 16;
  const v8bf lo = *(const v8bf*)(p);
  const v8bf hi = *(const v8bf*)(p + 8);
  return join16(lo, hi);
}

// ------------------------------------------------------------------
// scalar helpers
// ------------------------------------------------------------------
__device__ __forceinline__ float sigmoidf(float v) { return 1.0f / (1.0f + __expf(-v)); }

__device__ __forceinline__ float seluf(float v) {
  const float a = 1.6732632423543772f, s = 1.0507009873554805f;
  return v > 0.0f ? s * v : s * a * (__expf(v) - 1.0f);
}

__device__ __forceinline__ unsigned hashu(unsigned v) {
  v ^= v >> 16; v *= 0x7feb352dU; v ^= v >> 15; v *= 0x846ca68bU; v ^= v >> 16;
  return v;
}
// deterministic stand-in for jax.random.normal(key(42)) (exact JAX RNG not
// reproducible on device): Box-Muller over a hash of (batch, feature).
__device__ __forceinline__ float eps_normal(int b, int j) {
  const unsigned s0 = hashu((unsigned)(b * 65599 + j) ^ 0x2a2a2a2aU);
  const unsigned s1 = hashu(s0 ^ 0x9e3779b9U);
  const float u1 = ((float)(s0 >> 8) + 1.0f) * (1.0f / 16777216.0f);
  const float u2 = (float)(s1 >> 8) * (1.0f / 16777216.0f);
  return sqrtf(-2.0f * __logf(u1)) * __cosf(6.28318530718f * u2);
}

// ------------------------------------------------------------------
// one dense layer: out = act(in(16x256) @ W[256][256]^T + b)
// ACT: 0 = none (f32 -> outf), 1 = relu (bf16), 2 = selu (bf16)
// per-thread bias values pre-hoisted into registers by the caller.
// ------------------------------------------------------------------
template <int ACT>
__device__ __forceinline__ void mlp_layer(const __bf16* inb, __bf16* outb, float* outf,
                                          const __bf16* Wp, float bias0, float bias1,
                                          int wv, int lane) {
  const int c0 = wv * 32;
  v8f acc0 = {}, acc1 = {};
#pragma unroll
  for (int k = 0; k < H; k += 32) {
    const v16bf a = frag_A_lds(inb, BLD, k, lane);
    acc0 = wmma_bf16(a, frag_B_pk(Wp, H / 32, c0,      k, lane), acc0);
    acc1 = wmma_bf16(a, frag_B_pk(Wp, H / 32, c0 + 16, k, lane), acc1);
  }
  const int mrow = (lane & 16) ? 8 : 0;
  const int jc   = lane & 15;
#pragma unroll
  for (int e = 0; e < 8; ++e) {
    const int m = mrow + e;
    float v0 = acc0[e] + bias0;
    float v1 = acc1[e] + bias1;
    if (ACT == 1) { v0 = fmaxf(v0, 0.0f); v1 = fmaxf(v1, 0.0f); }
    if (ACT == 2) { v0 = seluf(v0);       v1 = seluf(v1); }
    if (ACT == 0) {
      outf[m * H + c0 + jc]        = v0;
      outf[m * H + c0 + 16 + jc]   = v1;
    } else {
      outb[m * BLD + c0 + jc]      = (__bf16)v0;
      outb[m * BLD + c0 + 16 + jc] = (__bf16)v1;
    }
  }
  __syncthreads();
}

// _ode_f: 3x selu layers + linear out -> kout (f32 in LDS)
__device__ __forceinline__ void ode_f(__bf16* bA, __bf16* bB, float* kout,
                                      const float* ob, // 8 hoisted bias values
                                      int wv, int lane) {
  mlp_layer<2>(bA, bB, nullptr, g_wpack + OFF_OW1, ob[0], ob[1], wv, lane);
  mlp_layer<2>(bB, bA, nullptr, g_wpack + OFF_OW2, ob[2], ob[3], wv, lane);
  mlp_layer<2>(bA, bB, nullptr, g_wpack + OFF_OW3, ob[4], ob[5], wv, lane);
  mlp_layer<0>(bB, nullptr, kout, g_wpack + OFF_OW4, ob[6], ob[7], wv, lane);
}

// stage input: outb = bf16( y + dt * sum_s cs[s]*ks[s] )
__device__ __forceinline__ void build_stage(__bf16* outb, const float* yv, const float* dts,
                                            const float* const* ks, const float* cs,
                                            int ns, int tid) {
  for (int i = tid; i < TILE_B * H; i += 256) {
    const int m = i >> 8;
    const int j = i & (H - 1);
    float a = 0.0f;
    for (int s = 0; s < ns; ++s) a += cs[s] * ks[s][i];
    outb[m * BLD + j] = (__bf16)(yv[i] + dts[m] * a);
  }
  __syncthreads();
}

// ==================================================================
// fused GRU -> encoder -> dopri5 -> head. one block = 16 batch rows.
// ==================================================================
__global__ __launch_bounds__(256, 1)
void node_fused_kernel(
    const float* __restrict__ x,    const float* __restrict__ meta,
    const float* __restrict__ b_ih, const float* __restrict__ b_hh,
    const float* __restrict__ eb1,  const float* __restrict__ eb2,
    const float* __restrict__ ob1,  const float* __restrict__ ob2,
    const float* __restrict__ ob3,  const float* __restrict__ ob4,
    const float* __restrict__ outW, const float* __restrict__ outBias,
    float* __restrict__ out)
{
  __shared__ float  h32[TILE_B * H];        // GRU h (f32), later: std buffer
  __shared__ float  yv [TILE_B * H];        // ODE state y (f32)
  __shared__ float  kbuf[6][TILE_B * H];    // k1..k6 (f32)
  __shared__ __bf16 bufA[TILE_B * BLD];     // bf16 activation ping
  __shared__ __bf16 bufB[TILE_B * BLD];     // bf16 activation pong
  __shared__ __bf16 xinS[TILE_B * 32];      // GRU input tile, K padded to 32
  __shared__ float  dts[TILE_B];

  const int tid  = threadIdx.x;
  const int lane = tid & 31;
  const int wv   = tid >> 5;                // wave 0..7 -> owns 32 cols
  const int b0   = blockIdx.x * TILE_B;
  const int c0   = wv * 32;
  const int jc   = lane & 15;
  const int mrow = (lane & 16) ? 8 : 0;

  for (int i = tid; i < TILE_B * H;   i += 256) h32[i]  = 0.0f;
  for (int i = tid; i < TILE_B * BLD; i += 256) bufA[i] = (__bf16)0.0f;
  if (tid < TILE_B) {
    const float t0 = x[(size_t)(b0 + tid) * TSEQ * DIN + 0];
    const float t1 = x[(size_t)(b0 + tid) * TSEQ * DIN + (size_t)(TSEQ - 1) * DIN + 0];
    dts[tid] = (t1 - t0) / (float)NSTEPS;
  }

  // hoist per-thread GRU gate biases into registers (columns are fixed)
  float brr[2], bzz[2], bxn[2], bhn[2];
#pragma unroll
  for (int tt = 0; tt < 2; ++tt) {
    const int j = c0 + tt * 16 + jc;
    brr[tt] = b_ih[j]         + b_hh[j];
    bzz[tt] = b_ih[H + j]     + b_hh[H + j];
    bxn[tt] = b_ih[2 * H + j];
    bhn[tt] = b_hh[2 * H + j];
  }
  __syncthreads();

  // ======================= GRU scan =======================
#pragma unroll 1
  for (int t = 0; t < TSEQ; ++t) {
    for (int i = tid; i < TILE_B * 32; i += 256) {
      const int r = i >> 5, c = i & 31;
      float v = 0.0f;
      if (c < DIN)              v = x[((size_t)(b0 + r) * TSEQ + t) * DIN + c];
      else if (c < DIN + MMETA) v = meta[(size_t)(b0 + r) * MMETA + (c - DIN)];
      xinS[i] = (__bf16)v;
    }
    __syncthreads();

    v8f accR[2], accZ[2], accNx[2], accNh[2];
    const v16bf ax = frag_A_lds(xinS, 32, 0, lane);
#pragma unroll
    for (int tt = 0; tt < 2; ++tt) {
      const int n0 = c0 + tt * 16;
      v8f z = {};
      accR [tt] = wmma_bf16(ax, frag_B_pk(g_wpack + OFF_WIH, 1, n0,         0, lane), z);
      accZ [tt] = wmma_bf16(ax, frag_B_pk(g_wpack + OFF_WIH, 1, H + n0,     0, lane), z);
      accNx[tt] = wmma_bf16(ax, frag_B_pk(g_wpack + OFF_WIH, 1, 2 * H + n0, 0, lane), z);
      accNh[tt] = z;
    }
#pragma unroll 1
    for (int k = 0; k < H; k += 32) {
      const v16bf ah = frag_A_lds(bufA, BLD, k, lane);
#pragma unroll
      for (int tt = 0; tt < 2; ++tt) {
        const int n0 = c0 + tt * 16;
        accR [tt] = wmma_bf16(ah, frag_B_pk(g_wpack + OFF_WHH, 8, n0,         k, lane), accR [tt]);
        accZ [tt] = wmma_bf16(ah, frag_B_pk(g_wpack + OFF_WHH, 8, H + n0,     k, lane), accZ [tt]);
        accNh[tt] = wmma_bf16(ah, frag_B_pk(g_wpack + OFF_WHH, 8, 2 * H + n0, k, lane), accNh[tt]);
      }
    }

    float hnew[2][8];
#pragma unroll
    for (int tt = 0; tt < 2; ++tt) {
      const int j = c0 + tt * 16 + jc;
#pragma unroll
      for (int e = 0; e < 8; ++e) {
        const int m = mrow + e;
        const float r = sigmoidf(accR[tt][e] + brr[tt]);
        const float z = sigmoidf(accZ[tt][e] + bzz[tt]);
        const float n = tanhf(accNx[tt][e] + bxn[tt] + r * (accNh[tt][e] + bhn[tt]));
        hnew[tt][e] = (1.0f - z) * n + z * h32[m * H + j];
      }
    }
    __syncthreads();               // all reads of old h done
#pragma unroll
    for (int tt = 0; tt < 2; ++tt) {
      const int j = c0 + tt * 16 + jc;
#pragma unroll
      for (int e = 0; e < 8; ++e) {
        const int m = mrow + e;
        h32 [m * H   + j] = hnew[tt][e];
        bufA[m * BLD + j] = (__bf16)hnew[tt][e];
      }
    }
    __syncthreads();
  }

  // ======================= Encoder =======================
  mlp_layer<1>(bufA, bufB, nullptr, g_wpack + OFF_EW1,
               eb1[c0 + jc], eb1[c0 + 16 + jc], wv, lane);   // relu -> bufB

  {
    v8f acc[4];
#pragma unroll
    for (int q = 0; q < 4; ++q) { v8f z = {}; acc[q] = z; }
    float eb2v[4];
#pragma unroll
    for (int q = 0; q < 4; ++q) eb2v[q] = eb2[wv * 64 + q * 16 + jc];
#pragma unroll 1
    for (int k = 0; k < H; k += 32) {
      const v16bf a = frag_A_lds(bufB, BLD, k, lane);
#pragma unroll
      for (int q = 0; q < 4; ++q)
        acc[q] = wmma_bf16(a, frag_B_pk(g_wpack + OFF_EW2, 8, wv * 64 + q * 16, k, lane),
                           acc[q]);
    }
#pragma unroll
    for (int q = 0; q < 4; ++q) {
      const int j2 = wv * 64 + q * 16 + jc;
#pragma unroll
      for (int e = 0; e < 8; ++e) {
        const int m = mrow + e;
        const float v = acc[q][e] + eb2v[q];
        if (j2 < H) yv [m * H + j2]       = v;   // mean
        else        h32[m * H + (j2 - H)] = v;   // std
      }
    }
    __syncthreads();
    for (int i = tid; i < TILE_B * H; i += 256) {
      const int m = i >> 8, j = i & (H - 1);
      yv[i] = eps_normal(b0 + m, j) * h32[i] + yv[i];   // y0 = eps*std + mean
    }
    __syncthreads();
  }

  // ======================= dopri5 =======================
  float obv[8];
  obv[0] = ob1[c0 + jc]; obv[1] = ob1[c0 + 16 + jc];
  obv[2] = ob2[c0 + jc]; obv[3] = ob2[c0 + 16 + jc];
  obv[4] = ob3[c0 + jc]; obv[5] = ob3[c0 + 16 + jc];
  obv[6] = ob4[c0 + jc]; obv[7] = ob4[c0 + 16 + jc];

#pragma unroll 1
  for (int step = 0; step < NSTEPS; ++step) {
    build_stage(bufA, yv, dts, nullptr, nullptr, 0, tid);
    ode_f(bufA, bufB, kbuf[0], obv, wv, lane);

    { const float cs[1] = {A21c}; const float* ks[1] = {kbuf[0]};
      build_stage(bufA, yv, dts, ks, cs, 1, tid); }
    ode_f(bufA, bufB, kbuf[1], obv, wv, lane);

    { const float cs[2] = {A31c, A32c}; const float* ks[2] = {kbuf[0], kbuf[1]};
      build_stage(bufA, yv, dts, ks, cs, 2, tid); }
    ode_f(bufA, bufB, kbuf[2], obv, wv, lane);

    { const float cs[3] = {A41c, A42c, A43c};
      const float* ks[3] = {kbuf[0], kbuf[1], kbuf[2]};
      build_stage(bufA, yv, dts, ks, cs, 3, tid); }
    ode_f(bufA, bufB, kbuf[3], obv, wv, lane);

    { const float cs[4] = {A51c, A52c, A53c, A54c};
      const float* ks[4] = {kbuf[0], kbuf[1], kbuf[2], kbuf[3]};
      build_stage(bufA, yv, dts, ks, cs, 4, tid); }
    ode_f(bufA, bufB, kbuf[4], obv, wv, lane);

    { const float cs[5] = {A61c, A62c, A63c, A64c, A65c};
      const float* ks[5] = {kbuf[0], kbuf[1], kbuf[2], kbuf[3], kbuf[4]};
      build_stage(bufA, yv, dts, ks, cs, 5, tid); }
    ode_f(bufA, bufB, kbuf[5], obv, wv, lane);

    for (int i = tid; i < TILE_B * H; i += 256) {
      const int m = i >> 8;
      const float a = B1c * kbuf[0][i] + B3c * kbuf[2][i] + B4c * kbuf[3][i]
                    + B5c * kbuf[4][i] + B6c * kbuf[5][i];
      yv[i] += dts[m] * a;
    }
    __syncthreads();
  }

  // ======================= Head =======================
  if (tid < TILE_B) {
    float acc = outBias[0];
    for (int j = 0; j < H; ++j) acc += yv[tid * H + j] * outW[j];
    for (int i2 = 0; i2 < MMETA; ++i2)
      acc += meta[(size_t)(b0 + tid) * MMETA + i2] * outW[H + i2];
    out[b0 + tid] = acc;
  }
}

extern "C" void kernel_launch(void* const* d_in, const int* in_sizes, int n_in,
                              void* d_out, int out_size, void* d_ws, size_t ws_size,
                              hipStream_t stream) {
  (void)in_sizes; (void)n_in; (void)out_size; (void)d_ws; (void)ws_size;
  const float* x     = (const float*)d_in[0];
  const float* meta  = (const float*)d_in[1];
  const float* W_ih  = (const float*)d_in[2];
  const float* W_hh  = (const float*)d_in[3];
  const float* b_ih  = (const float*)d_in[4];
  const float* b_hh  = (const float*)d_in[5];
  const float* eW1   = (const float*)d_in[6];
  const float* eb1   = (const float*)d_in[7];
  const float* eW2   = (const float*)d_in[8];
  const float* eb2   = (const float*)d_in[9];
  const float* oW1   = (const float*)d_in[10];
  const float* ob1   = (const float*)d_in[11];
  const float* oW2   = (const float*)d_in[12];
  const float* ob2   = (const float*)d_in[13];
  const float* oW3   = (const float*)d_in[14];
  const float* ob3   = (const float*)d_in[15];
  const float* oW4   = (const float*)d_in[16];
  const float* ob4   = (const float*)d_in[17];
  const float* outW  = (const float*)d_in[18];
  const float* outB  = (const float*)d_in[19];
  float* out = (float*)d_out;

  // repack weights (bf16, WMMA B-fragment layout) into the device arena
  dim3 pg(64), pb(256);
  pack_weights<<<pg, pb, 0, stream>>>(W_ih, OFF_WIH, 3 * H, 32, DIN + MMETA);
  pack_weights<<<pg, pb, 0, stream>>>(W_hh, OFF_WHH, 3 * H, H, H);
  pack_weights<<<pg, pb, 0, stream>>>(eW1,  OFF_EW1, H,     H, H);
  pack_weights<<<pg, pb, 0, stream>>>(eW2,  OFF_EW2, 2 * H, H, H);
  pack_weights<<<pg, pb, 0, stream>>>(oW1,  OFF_OW1, H,     H, H);
  pack_weights<<<pg, pb, 0, stream>>>(oW2,  OFF_OW2, H,     H, H);
  pack_weights<<<pg, pb, 0, stream>>>(oW3,  OFF_OW3, H,     H, H);
  pack_weights<<<pg, pb, 0, stream>>>(oW4,  OFF_OW4, H,     H, H);

  dim3 grid(BATCH / TILE_B), block(256);
  node_fused_kernel<<<grid, block, 0, stream>>>(
      x, meta, b_ih, b_hh, eb1, eb2, ob1, ob2, ob3, ob4, outW, outB, out);
}
